// AFTConv_74208444940328
// MI455X (gfx1250) — matrix operand order repaired and checked by reading, not compiled, not run on hardware
//
#include <hip/hip_runtime.h>
#include <hip/hip_bf16.h>

typedef __attribute__((ext_vector_type(16))) _Float16 v16h;
typedef __attribute__((ext_vector_type(8)))  float    v8f;
typedef __attribute__((ext_vector_type(4)))  float    f4;
typedef __attribute__((ext_vector_type(4)))  _Float16 h4;

#define HEADS 8
#define DHEAD 64
#define HH 48
#define WW 48
#define NPOS (HH*WW)          // 2304
#define BATCH 32
#define MROWS (BATCH*NPOS)    // 73728
#define CDIM 512
#define WIN 11
#define WELEMS (HEADS*WIN*WIN) // 968
#define EPS 1e-8f

// ---------------------------------------------------------------------------
// CDNA5 async global->LDS copy (16B per lane), tracked by ASYNCcnt.
// lds_addr: LDS byte offset (low 32 bits of generic pointer to __shared__).
// ---------------------------------------------------------------------------
__device__ __forceinline__ void async_copy_b128(unsigned lds_addr, const void* gaddr) {
  asm volatile("global_load_async_to_lds_b128 %0, %1, off"
               :: "v"(lds_addr), "v"(gaddr) : "memory");
}
__device__ __forceinline__ void wait_async0() {
  asm volatile("s_wait_asynccnt 0x0" ::: "memory");
}
__device__ __forceinline__ unsigned lds_off(const void* p) {
  return (unsigned)(unsigned long long)p;   // low 32 bits = LDS byte offset
}

// ---------------------------------------------------------------------------
// Kernel: fp32 -> f16 bulk convert (x)
// ---------------------------------------------------------------------------
__global__ __launch_bounds__(256) void cvt_x(const float* __restrict__ in,
                                             _Float16* __restrict__ out,
                                             size_t n4) {
  size_t i = (size_t)blockIdx.x * 256 + threadIdx.x;
  if (i < n4) {
    f4 v = ((const f4*)in)[i];
    h4 h;
    h.x = (_Float16)v.x; h.y = (_Float16)v.y;
    h.z = (_Float16)v.z; h.w = (_Float16)v.w;
    ((h4*)out)[i] = h;
  }
}

// ---------------------------------------------------------------------------
// Kernel: weight transpose + convert:  out[n][k] = in[k][n]  (f16),
// zero-pad rows n >= N (for WK padded 8 -> 16).
// ---------------------------------------------------------------------------
__global__ __launch_bounds__(256) void cvt_wt(const float* __restrict__ in,
                                              _Float16* __restrict__ out,
                                              int N, int Npad) {
  int idx = blockIdx.x * 256 + threadIdx.x;   // over Npad*512
  if (idx >= Npad * CDIM) return;
  int n = idx >> 9, k = idx & (CDIM - 1);
  out[idx] = (n < N) ? (_Float16)in[(size_t)k * N + n] : (_Float16)0.f;
}

// ---------------------------------------------------------------------------
// Kernel: BatchNorm2d(1) over conv weight w (8,1,11,11) + exp(.)-1
// ---------------------------------------------------------------------------
__global__ __launch_bounds__(256) void prep_w(const float* __restrict__ w,
                                              const float* __restrict__ gamma,
                                              const float* __restrict__ beta,
                                              float* __restrict__ wnorm) {
  __shared__ float s1[256], s2[256];
  int tid = threadIdx.x;
  float a = 0.f, b2 = 0.f;
  for (int i = tid; i < WELEMS; i += 256) { float v = w[i]; a += v; b2 += v * v; }
  s1[tid] = a; s2[tid] = b2;
  __syncthreads();
  for (int s = 128; s > 0; s >>= 1) {
    if (tid < s) { s1[tid] += s1[tid + s]; s2[tid] += s2[tid + s]; }
    __syncthreads();
  }
  float mean = s1[0] / (float)WELEMS;
  float var  = s2[0] / (float)WELEMS - mean * mean;
  float g = gamma[0], bb = beta[0];
  float rs = rsqrtf(var + EPS);
  for (int i = tid; i < WELEMS; i += 256) {
    float wn = g * (w[i] - mean) * rs + bb;
    wnorm[i] = __expf(wn) - 1.f;
  }
}

// ---------------------------------------------------------------------------
// Main GEMM: C[M,N] = A[M,K] @ Bt[N,K]^T, A/Bt f16, C fp32.
//   mode 0: sigmoid   mode 1: copy   mode 2: + bias[col]
//   Block tile 128x128, BK=32. 8 waves, each 32x64 = 2x4 WMMA tiles.
//   Ping-pong LDS, staged with global_load_async_to_lds_b128.
// ---------------------------------------------------------------------------
#define BM 128
#define BN 128
#define BK 32

__global__ __launch_bounds__(256) void gemm_f16(const _Float16* __restrict__ A,
                                                const _Float16* __restrict__ Bt,
                                                float* __restrict__ C,
                                                const float* __restrict__ bias,
                                                int M, int N, int K, int mode) {
  __shared__ _Float16 As[2][BM][BK];   // 2 x 8 KB
  __shared__ _Float16 Bs[2][BN][BK];   // 2 x 8 KB

  const int tid  = threadIdx.x;
  const int lane = tid & 31;
  const int wid  = tid >> 5;        // 0..7
  const int wm   = wid >> 1;        // 0..3 -> 32-row strip
  const int wn   = wid & 1;         // 0..1 -> 64-col strip
  const int row0 = blockIdx.x * BM;
  const int col0 = blockIdx.y * BN;

  const int half = lane >> 4;       // 0/1
  const int lrow = lane & 15;

  const unsigned asB[2] = { lds_off(&As[0][0][0]), lds_off(&As[1][0][0]) };
  const unsigned bsB[2] = { lds_off(&Bs[0][0][0]), lds_off(&Bs[1][0][0]) };

  v8f acc[2][4] = {};

  const int niter = K / BK;

  // ---- async stage of one k-slice into buffer `buf` ----
  auto stage = [&](int buf, int k0) {
    // A: 128 rows x 32 halves = 512 x 16B units, 2 per thread
    #pragma unroll
    for (int i = 0; i < 2; ++i) {
      int u = i * 256 + tid;
      int m = u >> 2, q = u & 3;
      async_copy_b128(asB[buf] + (unsigned)(m * BK + q * 8) * 2,
                      A + (size_t)(row0 + m) * K + k0 + q * 8);
    }
    // B: 128 n-rows x 32 halves = 512 x 16B units, 2 per thread
    #pragma unroll
    for (int i = 0; i < 2; ++i) {
      int u = i * 256 + tid;
      int n = u >> 2, q = u & 3;
      async_copy_b128(bsB[buf] + (unsigned)(n * BK + q * 8) * 2,
                      Bt + (size_t)(col0 + n) * K + k0 + q * 8);
    }
  };

  stage(0, 0);

  for (int it = 0; it < niter; ++it) {
    const int buf = it & 1;
    wait_async0();
    __syncthreads();
    if (it + 1 < niter) stage(buf ^ 1, (it + 1) * BK);

    // ---- fragments per ISA 7.12.2 16-bit layouts ----
    // A 16x32: lane(lrow,half): e0..7 -> K = half*8 + e ; e8..15 -> K = 16 + half*8 + e
    // B 32x16: lane(n=lrow,half): e0..7 -> K = half*8 + e ; e8..15 -> K = 16 + half*8 + e
    v16h af[2], bf[4];
    #pragma unroll
    for (int t = 0; t < 2; ++t) {
      int m = wm * 32 + t * 16 + lrow;
      #pragma unroll
      for (int e = 0; e < 8; ++e) af[t][e]     = As[buf][m][half * 8 + e];
      #pragma unroll
      for (int e = 0; e < 8; ++e) af[t][8 + e] = As[buf][m][16 + half * 8 + e];
    }
    #pragma unroll
    for (int t = 0; t < 4; ++t) {
      int n = wn * 64 + t * 16 + lrow;
      #pragma unroll
      for (int e = 0; e < 16; ++e) bf[t][e] = Bs[buf][n][(e >> 3) * 16 + half * 8 + (e & 7)];
    }
    #pragma unroll
    for (int tm = 0; tm < 2; ++tm)
      #pragma unroll
      for (int tn = 0; tn < 4; ++tn)
        acc[tm][tn] = __builtin_amdgcn_wmma_f32_16x16x32_f16(
            false, af[tm], false, bf[tn], (short)0, acc[tm][tn], false, false);
  }

  // ---- epilogue: C 16x16 f32 layout: VGPR i, lane -> row = 8*half + i, col = lrow ----
  #pragma unroll
  for (int tm = 0; tm < 2; ++tm) {
    #pragma unroll
    for (int tn = 0; tn < 4; ++tn) {
      int col = col0 + wn * 64 + tn * 16 + lrow;
      float bv = (mode == 2) ? bias[col] : 0.f;
      #pragma unroll
      for (int i = 0; i < 8; ++i) {
        int row = row0 + wm * 32 + tm * 16 + half * 8 + i;
        float v = acc[tm][tn][i];
        if (mode == 0)      v = 1.f / (1.f + __expf(-v));
        else if (mode == 2) v += bv;
        C[(size_t)row * N + col] = v;
      }
    }
  }
}

// ---------------------------------------------------------------------------
// k = x @ WK : A f16 [M,512], Bt f16 [16,512] (rows 8..15 zero), out fp32 [M,8]
// ---------------------------------------------------------------------------
__global__ __launch_bounds__(256) void gemm_k(const _Float16* __restrict__ A,
                                              const _Float16* __restrict__ Bt,
                                              float* __restrict__ Kout) {
  __shared__ _Float16 As[2][128][32];
  __shared__ _Float16 Bs[2][16][32];
  const int tid = threadIdx.x, lane = tid & 31, wid = tid >> 5;
  const int row0 = blockIdx.x * 128;
  const int half = lane >> 4, lrow = lane & 15;
  const unsigned asB[2] = { lds_off(&As[0][0][0]), lds_off(&As[1][0][0]) };
  const unsigned bsB[2] = { lds_off(&Bs[0][0][0]), lds_off(&Bs[1][0][0]) };
  v8f acc = {};

  auto stage = [&](int buf, int k0) {
    #pragma unroll
    for (int i = 0; i < 2; ++i) {
      int u = i * 256 + tid;
      int m = u >> 2, q = u & 3;
      async_copy_b128(asB[buf] + (unsigned)(m * 32 + q * 8) * 2,
                      A + (size_t)(row0 + m) * CDIM + k0 + q * 8);
    }
    if (tid < 64) {
      int n = tid >> 2, q = tid & 3;
      async_copy_b128(bsB[buf] + (unsigned)(n * 32 + q * 8) * 2,
                      Bt + (size_t)n * CDIM + k0 + q * 8);
    }
  };

  stage(0, 0);
  const int niter = CDIM / 32;
  for (int it = 0; it < niter; ++it) {
    const int buf = it & 1;
    wait_async0();
    __syncthreads();
    if (it + 1 < niter) stage(buf ^ 1, (it + 1) * 32);

    v16h af, bf;
    int m = wid * 16 + lrow;
    #pragma unroll
    for (int e = 0; e < 8; ++e) af[e]     = As[buf][m][half * 8 + e];
    #pragma unroll
    for (int e = 0; e < 8; ++e) af[8 + e] = As[buf][m][16 + half * 8 + e];
    #pragma unroll
    for (int e = 0; e < 16; ++e) bf[e]    = Bs[buf][lrow][(e >> 3) * 16 + half * 8 + (e & 7)];
    acc = __builtin_amdgcn_wmma_f32_16x16x32_f16(false, af, false, bf,
                                                 (short)0, acc, false, false);
  }
  if (lrow < HEADS) {
    #pragma unroll
    for (int i = 0; i < 8; ++i) {
      int row = row0 + wid * 16 + half * 8 + i;
      Kout[(size_t)row * HEADS + lrow] = acc[i];
    }
  }
}

// ---------------------------------------------------------------------------
// Global context: gc[b,p,d] = sum_n exp(k)*v ; gn[b,p] = sum_n exp(k)
// ---------------------------------------------------------------------------
__global__ __launch_bounds__(256) void reduce_gc(const float* __restrict__ kbuf,
                                                 const float* __restrict__ vbuf,
                                                 float* __restrict__ gc,
                                                 float* __restrict__ gn) {
  int p = blockIdx.x;
  int b = blockIdx.y;
  int tid = threadIdx.x;
  int d = tid & 63, nn = tid >> 6;
  const size_t baseN = (size_t)b * NPOS;
  float accv = 0.f, accs = 0.f;
  for (int n = nn; n < NPOS; n += 4) {
    float ke = __expf(kbuf[(baseN + n) * HEADS + p]);
    accv += ke * vbuf[(baseN + n) * CDIM + p * DHEAD + d];
    accs += ke;
  }
  __shared__ float sv[256];
  __shared__ float ss[4];
  sv[tid] = accv;
  if (d == 0) ss[nn] = accs;
  __syncthreads();
  if (nn == 0) {
    float s = sv[d] + sv[64 + d] + sv[128 + d] + sv[192 + d];
    gc[((size_t)b * HEADS + p) * DHEAD + d] = s;
    if (d == 0) gn[b * HEADS + p] = ss[0] + ss[1] + ss[2] + ss[3];
  }
}

// ---------------------------------------------------------------------------
// Depthwise 11x11 conv of (k*v) and k, combine with gc/gn/q -> outs (f16).
//   Block: 8x8 spatial tile x 32 channels of one head.  LDS halo 18x18.
// ---------------------------------------------------------------------------
#define CT 8
#define HALO 18

__global__ __launch_bounds__(256) void aft_conv(const float* __restrict__ kbuf,
                                                const float* __restrict__ vbuf,
                                                const float* __restrict__ qbuf,
                                                const float* __restrict__ gc,
                                                const float* __restrict__ gn,
                                                const float* __restrict__ wnorm,
                                                _Float16* __restrict__ outs) {
  int tileId = blockIdx.x;
  int ty = tileId / (WW / CT), tx = tileId % (WW / CT);
  int p = blockIdx.y >> 1;
  int ch0 = (blockIdx.y & 1) * 32;
  int b = blockIdx.z;
  int tid = threadIdx.x;
  int c = tid & 31, s = tid >> 5;

  __shared__ float vsh[HALO * HALO * 32];
  __shared__ float ksh[HALO * HALO];
  __shared__ float wsh[WIN * WIN];

  const size_t baseN = (size_t)b * NPOS;
  const int y0 = ty * CT - WIN / 2;
  const int x0 = tx * CT - WIN / 2;

  for (int idx = tid; idx < HALO * HALO; idx += 256) {
    int ly = idx / HALO, lx = idx % HALO;
    int gy = y0 + ly, gx = x0 + lx;
    float kv = 0.f;
    if (gy >= 0 && gy < HH && gx >= 0 && gx < WW)
      kv = kbuf[(baseN + gy * WW + gx) * HEADS + p];
    ksh[idx] = kv;
  }
  for (int idx = tid; idx < HALO * HALO * 32; idx += 256) {
    int cc = idx & 31;
    int sp = idx >> 5;
    int ly = sp / HALO, lx = sp % HALO;
    int gy = y0 + ly, gx = x0 + lx;
    float vv = 0.f;
    if (gy >= 0 && gy < HH && gx >= 0 && gx < WW)
      vv = vbuf[(baseN + gy * WW + gx) * CDIM + p * DHEAD + ch0 + cc];
    vsh[idx] = vv;
  }
  if (tid < WIN * WIN) wsh[tid] = wnorm[p * WIN * WIN + tid];
  __syncthreads();

  const float gcv = gc[((size_t)b * HEADS + p) * DHEAD + ch0 + c];
  const float gnv = gn[b * HEADS + p];

  for (int ox = 0; ox < CT; ++ox) {
    float num = 0.f, den = 0.f;
    for (int dy = 0; dy < WIN; ++dy) {
      int rbase = (s + dy) * HALO + ox;
      #pragma unroll
      for (int dx = 0; dx < WIN; ++dx) {
        float wv = wsh[dy * WIN + dx];
        float kk = ksh[rbase + dx];
        float vv = vsh[(rbase + dx) * 32 + c];
        num = fmaf(wv * kk, vv, num);
        den = fmaf(wv, kk, den);
      }
    }
    int gy = ty * CT + s, gx = tx * CT + ox;
    size_t nidx = baseN + gy * WW + gx;
    size_t cidx = nidx * CDIM + p * DHEAD + ch0 + c;
    float qv = qbuf[cidx];
    outs[cidx] = (_Float16)(qv * ((num + gcv) / (den + gnv + EPS)));
  }
}

// ---------------------------------------------------------------------------
// Launch
// ---------------------------------------------------------------------------
extern "C" void kernel_launch(void* const* d_in, const int* in_sizes, int n_in,
                              void* d_out, int out_size, void* d_ws, size_t ws_size,
                              hipStream_t stream) {
  const float* x     = (const float*)d_in[0];
  const float* WQ    = (const float*)d_in[1];
  const float* WK    = (const float*)d_in[2];
  const float* WV    = (const float*)d_in[3];
  const float* Wout  = (const float*)d_in[4];
  const float* bout  = (const float*)d_in[5];
  const float* w     = (const float*)d_in[6];
  const float* gamma = (const float*)d_in[7];
  const float* beta  = (const float*)d_in[8];
  float* out = (float*)d_out;

  // workspace carve-up (byte offsets; all sizes 4KB-aligned chunks)
  char* p = (char*)d_ws;
  float* wnorm = (float*)p;           p += 4096;
  float* kbuf  = (float*)p;           p += (size_t)MROWS * HEADS * 4;
  float* gcbuf = (float*)p;           p += (size_t)BATCH * HEADS * DHEAD * 4;
  float* gnbuf = (float*)p;           p += 4096;
  float* qbuf  = (float*)p;           p += (size_t)MROWS * CDIM * 4;
  float* vbuf  = (float*)p;           p += (size_t)MROWS * CDIM * 4;
  _Float16* xh    = (_Float16*)p;     p += (size_t)MROWS * CDIM * 2;
  _Float16* oh    = (_Float16*)p;     p += (size_t)MROWS * CDIM * 2;
  _Float16* wqt   = (_Float16*)p;     p += (size_t)CDIM * CDIM * 2;
  _Float16* wvt   = (_Float16*)p;     p += (size_t)CDIM * CDIM * 2;
  _Float16* woutt = (_Float16*)p;     p += (size_t)CDIM * CDIM * 2;
  _Float16* wkt   = (_Float16*)p;     p += (size_t)16 * CDIM * 2;

  // converts
  size_t n4 = (size_t)MROWS * CDIM / 4;
  cvt_x<<<(unsigned)((n4 + 255) / 256), 256, 0, stream>>>(x, xh, n4);
  cvt_wt<<<(CDIM * CDIM) / 256, 256, 0, stream>>>(WQ, wqt, CDIM, CDIM);
  cvt_wt<<<(CDIM * CDIM) / 256, 256, 0, stream>>>(WV, wvt, CDIM, CDIM);
  cvt_wt<<<(CDIM * CDIM) / 256, 256, 0, stream>>>(Wout, woutt, CDIM, CDIM);
  cvt_wt<<<(16 * CDIM) / 256, 256, 0, stream>>>(WK, wkt, HEADS, 16);
  prep_w<<<1, 256, 0, stream>>>(w, gamma, beta, wnorm);

  dim3 gGemm(MROWS / BM, CDIM / BN);   // 576 x 4
  gemm_f16<<<gGemm, 256, 0, stream>>>(xh, wqt, qbuf, nullptr, MROWS, CDIM, CDIM, 0);
  gemm_f16<<<gGemm, 256, 0, stream>>>(xh, wvt, vbuf, nullptr, MROWS, CDIM, CDIM, 1);
  gemm_k<<<MROWS / 128, 256, 0, stream>>>(xh, wkt, kbuf);

  reduce_gc<<<dim3(HEADS, BATCH), 256, 0, stream>>>(kbuf, vbuf, gcbuf, gnbuf);

  aft_conv<<<dim3((HH / CT) * (WW / CT), HEADS * 2, BATCH), 256, 0, stream>>>(
      kbuf, vbuf, qbuf, gcbuf, gnbuf, wnorm, oh);

  gemm_f16<<<gGemm, 256, 0, stream>>>(oh, woutt, out, bout, MROWS, CDIM, CDIM, 2);
}
